// PiNet2Torch_5196910428402
// MI455X (gfx1250) — matrix operand order repaired and checked by reading, not compiled, not run on hardware
//
#include <hip/hip_runtime.h>
#include <math.h>

typedef __attribute__((ext_vector_type(16))) _Float16 v16h;
typedef __attribute__((ext_vector_type(8)))  float    v8f;

#define N_ATOMS 10000
#define N_PAIRS 320000

#define WMMA_F16(acc, a, b) \
  (acc) = __builtin_amdgcn_wmma_f32_16x16x32_f16(false, (a), false, (b), (short)0, (acc), false, false)

// Branch-free tanh: gfx1250 has V_TANH_F32; otherwise exp2+rcp (no exec-mask branches).
__device__ __forceinline__ float fast_tanh(float x) {
#if __has_builtin(__builtin_amdgcn_tanhf)
  return __builtin_amdgcn_tanhf(x);
#elif __has_builtin(__builtin_amdgcn_exp2f) && __has_builtin(__builtin_amdgcn_rcpf)
  float e = __builtin_amdgcn_exp2f(x * 2.8853900817779268f); // 2*log2(e)
  return 1.0f - 2.0f * __builtin_amdgcn_rcpf(e + 1.0f);
#else
  float e = exp2f(x * 2.8853900817779268f);
  return 1.0f - 2.0f / (e + 1.0f);
#endif
}

// ---- gather-style fragment builders (k_p1pre; frag_a_f32 also in k_pair stage 2) -------
__device__ __forceinline__ v16h frag_a_f16(const _Float16* base, int ldr, int k0, int lane) {
  int m = lane & 15, kb = (lane & 16) ? 8 : 0;
  const _Float16* row = base + m * ldr + k0;
  v16h a;
#pragma unroll
  for (int e = 0; e < 8; ++e) { a[e] = row[kb + e]; a[8 + e] = row[16 + kb + e]; }
  return a;
}
__device__ __forceinline__ v16h frag_a_f32(const float* base, int ldr, int k0, int lane) {
  int m = lane & 15, kb = (lane & 16) ? 8 : 0;
  const float* row = base + m * ldr + k0;
  v16h a;
#pragma unroll
  for (int e = 0; e < 8; ++e) { a[e] = (_Float16)row[kb + e]; a[8 + e] = (_Float16)row[16 + kb + e]; }
  return a;
}
__device__ __forceinline__ v16h frag_b(const _Float16* w, int ldb, int k0, int n0, int lane) {
  int n = lane & 15, kb = (lane & 16) ? 8 : 0;
  const _Float16* p = w + (k0 + kb) * ldb + n0 + n;
  v16h b;
#pragma unroll
  for (int e = 0; e < 16; ++e) b[e] = p[(e + (e >> 3) * 8) * ldb];
  return b;
}
// Swizzled (fragment-major) B: each lane's 16 halfs contiguous -> 2x ds_load_b128.
__device__ __forceinline__ v16h frag_b_sw(const _Float16* buf, int fragIdx, int lane) {
  return *(const v16h*)(buf + ((unsigned)fragIdx * 32u + (unsigned)lane) * 16u);
}
// A fragment straight from global p1_in (f16): i/j choice is uniform per kc,
// and each lane's data is two contiguous 16B runs -> 2x global_load_b128.
__device__ __forceinline__ v16h frag_a_pair(const _Float16* __restrict__ p1in16,
                                            const int* ijrow, int mt, int kc, int lane) {
  int m  = mt * 16 + (lane & 15);
  int kb = (lane & 16) ? 8 : 0;
  int atom = (kc < 2) ? ijrow[m * 2] : ijrow[m * 2 + 1];
  const uint4* base = (const uint4*)(p1in16 + atom * 64 + (kc & 1) * 32 + kb);
  union { uint4 u[2]; v16h v; } t;
  t.u[0] = base[0];   // halfs k0+kb .. +7
  t.u[1] = base[2];   // halfs k0+16+kb .. +7
  return t.v;
}

// ---------------------------------------------------------------------------
// Kernel 0: pre-swizzle pi_W into fragment-major f16 chunks in workspace.
// dst idx = (((ch*4+kc)*5+t)*32+lane)*16+e ; 8 chunks of 80 cols.
// ---------------------------------------------------------------------------
__global__ __launch_bounds__(256) void k_convW(
    const float* __restrict__ piW, _Float16* __restrict__ dst) {
  int idx = blockIdx.x * 256 + threadIdx.x;      // 81920 exactly
  int e = idx & 15, lane = (idx >> 4) & 31;
  int rest = idx >> 9;
  int t = rest % 5, rest2 = rest / 5;
  int kc = rest2 & 3, ch = rest2 >> 2;
  int kr = (lane >> 4) * 8 + e + (e >> 3) * 8;
  int k  = kc * 32 + kr;
  int n  = ch * 80 + t * 16 + (lane & 15);
  dst[idx] = (_Float16)piW[k * 640 + n];
}

// ---------------------------------------------------------------------------
// Kernel 1: p1_in = tanh(tanh(p1@W1+b1)@W2+b2) -> f16 (feeds pair GEMM A)
// ---------------------------------------------------------------------------
__global__ __launch_bounds__(256) void k_p1pre(
    const float* __restrict__ p1, const float* __restrict__ W1,
    const float* __restrict__ b1, const float* __restrict__ W2,
    const float* __restrict__ b2, _Float16* __restrict__ p1in16) {
  __shared__ _Float16 sW1[64 * 64];
  __shared__ _Float16 sW2[64 * 64];
  __shared__ float    sb1[64], sb2[64];
  __shared__ alignas(32) _Float16 sA[8][16 * 64];
  __shared__ alignas(32) _Float16 sAct[8][16 * 64];

  int tid = threadIdx.x, lane = tid & 31, w = tid >> 5;
  for (int i = tid; i < 64 * 64; i += 256) { sW1[i] = (_Float16)W1[i]; sW2[i] = (_Float16)W2[i]; }
  for (int i = tid; i < 64; i += 256)       { sb1[i] = b1[i]; sb2[i] = b2[i]; }

  int tile = blockIdx.x * 8 + w;
  bool active = tile < (N_ATOMS / 16);
  int atomBase = (active ? tile : 0) * 16;
  for (int i = lane; i < 16 * 64; i += 32)
    sA[w][i] = (_Float16)(active ? p1[(atomBase + (i >> 6)) * 64 + (i & 63)] : 0.0f);
  __syncthreads();

  v16h a0 = frag_a_f16(sA[w], 64, 0, lane);
  v16h a1 = frag_a_f16(sA[w], 64, 32, lane);
#pragma unroll
  for (int nt = 0; nt < 4; ++nt) {
    v8f acc = {};
    WMMA_F16(acc, a0, frag_b(sW1, 64, 0,  nt * 16, lane));
    WMMA_F16(acc, a1, frag_b(sW1, 64, 32, nt * 16, lane));
    int n = nt * 16 + (lane & 15);
#pragma unroll
    for (int r = 0; r < 8; ++r) {
      int m = r + ((lane & 16) ? 8 : 0);
      sAct[w][m * 64 + n] = (_Float16)fast_tanh(acc[r] + sb1[n]);
    }
  }
  __syncthreads();

  v16h c0 = frag_a_f16(sAct[w], 64, 0, lane);
  v16h c1 = frag_a_f16(sAct[w], 64, 32, lane);
#pragma unroll
  for (int nt = 0; nt < 4; ++nt) {
    v8f acc = {};
    WMMA_F16(acc, c0, frag_b(sW2, 64, 0,  nt * 16, lane));
    WMMA_F16(acc, c1, frag_b(sW2, 64, 32, nt * 16, lane));
    int n = nt * 16 + (lane & 15);
#pragma unroll
    for (int r = 0; r < 8; ++r) {
      int m = r + ((lane & 16) ? 8 : 0);
      if (active) p1in16[(atomBase + m) * 64 + n] = (_Float16)fast_tanh(acc[r] + sb2[n]);
    }
  }
}

// ---------------------------------------------------------------------------
// Kernel 2: v = segment_sum(d3 * fc_edge, i)
// ---------------------------------------------------------------------------
__global__ __launch_bounds__(256) void k_v(
    const int* __restrict__ ind2, const float* __restrict__ d3,
    const float* __restrict__ fc, float* __restrict__ v) {
  int p = blockIdx.x * 256 + threadIdx.x;
  int ia = ind2[p * 2];
  float f = fc[p];
#pragma unroll
  for (int x = 0; x < 3; ++x)
    __hip_atomic_fetch_add(&v[ia * 3 + x], d3[p * 3 + x] * f,
                           __ATOMIC_RELAXED, __HIP_MEMORY_SCOPE_AGENT);
}

// ---------------------------------------------------------------------------
// Kernel 3: fused pair pipeline. 32 pairs per wave (2 M-tiles share each B
// fragment); pi_W chunks software-pipelined through registers into LDS.
// 1250 blocks * 8 waves * 32 pairs = 320000 exactly.
// ---------------------------------------------------------------------------
__global__ __launch_bounds__(256) void k_pair(
    const int* __restrict__ ind2, const float* __restrict__ basis,
    const float* __restrict__ d3, const float* __restrict__ fc,
    const uint4* __restrict__ piWsw, const float* __restrict__ pib,
    const float* __restrict__ iiW, const float* __restrict__ p3,
    const _Float16* __restrict__ p1in16, const float* __restrict__ v,
    float* __restrict__ p1scat, float* __restrict__ p3scat) {
  __shared__ alignas(32) _Float16 sPiW[128 * 80];   // swizzled chunk (20 frag tiles)
  __shared__ float    sPib[640];
  __shared__ alignas(32) _Float16 sIiW[64 * 128];   // swizzled (16 frag tiles)
  __shared__ float    sIp[8][32 * 64];              // per-wave i_pair accumulator
  __shared__ float    sBasis[8][32 * 10];
  __shared__ float    sMeta[8][32 * 8];             // d3[3], t3[3], tb_gate
  __shared__ int      sIJ[8][32 * 2];

  int tid = threadIdx.x, lane = tid & 31, w = tid >> 5;
  int pBase = (blockIdx.x * 8 + w) * 32;

  // ii_W -> LDS, fragment-swizzled: idx = ((kc*8+nt)*32+lane)*16+e
  for (int idx = tid; idx < 64 * 128; idx += 256) {
    int e = idx & 15, ln = (idx >> 4) & 31, nt = (idx >> 9) & 7, kc = idx >> 12;
    int k = kc * 32 + (ln >> 4) * 8 + e + (e >> 3) * 8;
    int n = nt * 16 + (ln & 15);
    sIiW[idx] = (_Float16)iiW[k * 128 + n];
  }
  for (int i = tid; i < 640; i += 256) sPib[i] = pib[i];

  { // per-pair meta: lane handles pair m=lane (32 pairs/wave)
    int p  = pBase + lane;
    int ia = ind2[p * 2], ja = ind2[p * 2 + 1];
    sIJ[w][lane * 2] = ia; sIJ[w][lane * 2 + 1] = ja;
    float f = fc[p];
    float dx[3], vx[3], wv[3];
    float proj = 0.f;
#pragma unroll
    for (int x = 0; x < 3; ++x) { dx[x] = d3[p * 3 + x]; vx[x] = v[ia * 3 + x]; proj += vx[x] * dx[x]; }
    float w2 = 0.f;
#pragma unroll
    for (int x = 0; x < 3; ++x) { wv[x] = vx[x] - proj * dx[x]; w2 += wv[x] * wv[x]; }
    float g  = w2 / (w2 + 1e-4f);            // (10*EPS)^2
    float rs = rsqrtf(w2 + 1e-6f);           // EPS^2
#pragma unroll
    for (int x = 0; x < 3; ++x) { sMeta[w][lane * 8 + x] = dx[x]; sMeta[w][lane * 8 + 3 + x] = wv[x] * rs * g; }
    sMeta[w][lane * 8 + 6] = g * f * f;      // tb_gate
  }
  for (int i = lane; i < 320; i += 32)      sBasis[w][i] = basis[pBase * 10 + i];
  for (int i = lane; i < 32 * 64; i += 32)  sIp[w][i] = 0.f;
  __syncthreads();

  // A fragments (2 M-tiles x 4 K-chunks) straight from global, kept in VGPRs
  v16h aF[2][4];
#pragma unroll
  for (int mt = 0; mt < 2; ++mt)
#pragma unroll
    for (int kc = 0; kc < 4; ++kc) aF[mt][kc] = frag_a_pair(p1in16, sIJ[w], mt, kc, lane);

  // Stage 1: (32x128)@(128x640) + bias + tanh + basis contraction, fused.
  // Chunk ch+1 is loaded into registers while WMMAs run on chunk ch.
  uint4 st[5];
#pragma unroll
  for (int q = 0; q < 5; ++q) st[q] = piWsw[tid * 5 + q];
  for (int ch = 0; ch < 8; ++ch) {
#pragma unroll
    for (int q = 0; q < 5; ++q) ((uint4*)sPiW)[tid * 5 + q] = st[q];
    __syncthreads();
    if (ch < 7) {
#pragma unroll
      for (int q = 0; q < 5; ++q) st[q] = piWsw[(ch + 1) * 1280 + tid * 5 + q];
    }
    int c0 = ch * 80;
#pragma unroll
    for (int t = 0; t < 5; ++t) {
      v16h bF[4];
#pragma unroll
      for (int kc = 0; kc < 4; ++kc) bF[kc] = frag_b_sw(sPiW, kc * 5 + t, lane);
      int n = c0 + t * 16 + (lane & 15);
      int cch = n / 10, bidx = n % 10;
      float bias = sPib[n];
#pragma unroll
      for (int mt = 0; mt < 2; ++mt) {
        v8f acc = {};
#pragma unroll
        for (int kc = 0; kc < 4; ++kc) WMMA_F16(acc, aF[mt][kc], bF[kc]);
#pragma unroll
        for (int r = 0; r < 8; ++r) {
          int m = mt * 16 + r + ((lane & 16) ? 8 : 0);
          float hval = fast_tanh(acc[r] + bias);
          __hip_atomic_fetch_add(&sIp[w][m * 64 + cch], hval * sBasis[w][m * 10 + bidx],
                                 __ATOMIC_RELAXED, __HIP_MEMORY_SCOPE_WORKGROUP);
        }
      }
    }
    __syncthreads();
  }

  // Stage 2: tanh(i_pair @ ii_W) + fused equivariant scatter
#pragma unroll
  for (int mt = 0; mt < 2; ++mt) {
    v16h a2lo = frag_a_f32(sIp[w] + mt * 16 * 64, 64, 0, lane);
    v16h a2hi = frag_a_f32(sIp[w] + mt * 16 * 64, 64, 32, lane);
#pragma unroll
    for (int nt = 0; nt < 8; ++nt) {
      v8f acc = {};
      WMMA_F16(acc, a2lo, frag_b_sw(sIiW, 0 * 8 + nt, lane));
      WMMA_F16(acc, a2hi, frag_b_sw(sIiW, 1 * 8 + nt, lane));
      int n = nt * 16 + (lane & 15);
#pragma unroll
      for (int r = 0; r < 8; ++r) {
        int m = mt * 16 + r + ((lane & 16) ? 8 : 0);
        float t = fast_tanh(acc[r]);
        int ia = sIJ[w][m * 2];
        __hip_atomic_fetch_add(&p1scat[ia * 128 + n], t,
                               __ATOMIC_RELAXED, __HIP_MEMORY_SCOPE_AGENT);
        if (n >= 64) {                 // i1b channels drive the p3 update
          int cc = n - 64;
          int ja = sIJ[w][m * 2 + 1];
          float tb = sMeta[w][m * 8 + 6];
#pragma unroll
          for (int x = 0; x < 3; ++x) {
            float d3x = sMeta[w][m * 8 + x];
            float t3x = sMeta[w][m * 8 + 3 + x];
            float p3jv = p3[ja * 192 + x * 64 + cc];
            __hip_atomic_fetch_add(&p3scat[ia * 192 + x * 64 + cc],
                                   t * (p3jv + d3x + t3x * tb),
                                   __ATOMIC_RELAXED, __HIP_MEMORY_SCOPE_AGENT);
          }
        }
      }
    }
  }
}

// ------------------- small per-atom tail (< 1 GFLOP total) ------------------
__global__ __launch_bounds__(256) void k_post1(
    const float* __restrict__ p1scat, const float* __restrict__ W, float* __restrict__ o) {
  int g = blockIdx.x * 256 + threadIdx.x;  int a = g >> 6, c = g & 63;
  float s = 0.f;
  for (int k = 0; k < 128; ++k) s += p1scat[a * 128 + k] * W[k * 64 + c];
  o[g] = fast_tanh(s);
}
__global__ __launch_bounds__(256) void k_post2(
    const float* __restrict__ h, const float* __restrict__ W, float* __restrict__ o) {
  int g = blockIdx.x * 256 + threadIdx.x;  int a = g >> 6, c = g & 63;
  float s = 0.f;
  for (int k = 0; k < 64; ++k) s += h[a * 64 + k] * W[k * 64 + c];
  o[g] = fast_tanh(s);
}
__global__ __launch_bounds__(256) void k_eq(
    const float* __restrict__ p3scat, const float* __restrict__ W, float* __restrict__ o) {
  int g = blockIdx.x * 256 + threadIdx.x;
  int a = g / 192, rem = g % 192, x = rem >> 6, c = rem & 63;
  float s = 0.f;
  for (int k = 0; k < 64; ++k) s += p3scat[a * 192 + x * 64 + k] * W[k * 64 + c];
  o[g] = s;
}
__global__ __launch_bounds__(256) void k_h2a(
    const float* __restrict__ p1new, const float* __restrict__ p3new,
    const float* __restrict__ W, const float* __restrict__ b, float* __restrict__ o) {
  int g = blockIdx.x * 256 + threadIdx.x;  int a = g >> 6, c = g & 63;
  float s = b[c];
  for (int k = 0; k < 64; ++k) {
    float pk = p3new[a * 192 + k];
    float qk = p3new[a * 192 + 64 + k];
    float rk = p3new[a * 192 + 128 + k];
    float dotted = pk * pk + qk * qk + rk * rk;
    s += p1new[a * 64 + k] * W[k * 64 + c] + dotted * W[(64 + k) * 64 + c];
  }
  o[g] = fast_tanh(s);
}
__global__ __launch_bounds__(256) void k_final(
    const float* __restrict__ h2a, const float* __restrict__ p3new,
    const float* __restrict__ W, const float* __restrict__ b, float* __restrict__ out) {
  int g = blockIdx.x * 256 + threadIdx.x;  int a = g >> 7, n = g & 127;
  float s = b[n];
  for (int k = 0; k < 64; ++k) s += h2a[a * 64 + k] * W[k * 128 + n];
  float val = fast_tanh(s);
  if (n < 64) {
    out[a * 64 + n] = val;                                   // p1_out
  } else {
    int c = n - 64;
#pragma unroll
    for (int x = 0; x < 3; ++x)
      out[640000 + a * 192 + x * 64 + c] = p3new[a * 192 + x * 64 + c] * val;  // p3_out
  }
}

// ---------------------------------------------------------------------------
extern "C" void kernel_launch(void* const* d_in, const int* in_sizes, int n_in,
                              void* d_out, int out_size, void* d_ws, size_t ws_size,
                              hipStream_t stream) {
  (void)in_sizes; (void)n_in; (void)out_size; (void)ws_size;
  const int*   ind2  = (const int*)  d_in[0];
  const float* p1    = (const float*)d_in[1];
  const float* p3    = (const float*)d_in[2];
  const float* basis = (const float*)d_in[3];
  const float* d3    = (const float*)d_in[4];
  const float* fce   = (const float*)d_in[5];
  const float* preW1 = (const float*)d_in[6];
  const float* preb1 = (const float*)d_in[7];
  const float* preW2 = (const float*)d_in[8];
  const float* preb2 = (const float*)d_in[9];
  const float* piW   = (const float*)d_in[10];
  const float* pib   = (const float*)d_in[11];
  const float* iiW   = (const float*)d_in[12];
  const float* postW1= (const float*)d_in[13];
  const float* postW2= (const float*)d_in[14];
  const float* eqW   = (const float*)d_in[15];
  const float* pp1W1 = (const float*)d_in[16];
  const float* pp1b1 = (const float*)d_in[17];
  const float* pp1W2 = (const float*)d_in[18];
  const float* pp1b2 = (const float*)d_in[19];

  float* wsF     = (float*)d_ws;
  float* p1scat  = wsF;                 // 1,280,000 f
  float* p3scat  = wsF + 1280000;       // 1,920,000 f
  float* vbuf    = wsF + 3200000;       //    30,000 f
  float* p1new   = wsF + 3230000;       //   640,000 f
  float* htmp    = wsF + 3870000;       //   640,000 f
  float* p3new   = wsF + 4510000;       // 1,920,000 f
  float* h2a     = wsF + 6430000;       //   640,000 f
  _Float16* p1in16 = (_Float16*)(wsF + 7070000); // 640,000 halfs
  _Float16* piWsw  = (_Float16*)(wsF + 7390000); //  81,920 halfs (swizzled pi_W)

  // zero atomic accumulation buffers (p1scat + p3scat + v contiguous)
  hipMemsetAsync(p1scat, 0, (size_t)3230000 * sizeof(float), stream);

  k_convW<<<320,  256, 0, stream>>>(piW, piWsw);
  k_p1pre<<<79,   256, 0, stream>>>(p1, preW1, preb1, preW2, preb2, p1in16);
  k_v    <<<1250, 256, 0, stream>>>(ind2, d3, fce, vbuf);
  k_pair <<<1250, 256, 0, stream>>>(ind2, basis, d3, fce, (const uint4*)piWsw, pib, iiW, p3,
                                    p1in16, vbuf, p1scat, p3scat);
  k_post1<<<2500, 256, 0, stream>>>(p1scat, postW1, htmp);
  k_post2<<<2500, 256, 0, stream>>>(htmp, postW2, p1new);
  k_eq   <<<7500, 256, 0, stream>>>(p3scat, eqW, p3new);
  k_h2a  <<<2500, 256, 0, stream>>>(p1new, p3new, pp1W1, pp1b1, h2a);
  k_final<<<5000, 256, 0, stream>>>(h2a, p3new, pp1W2, pp1b2, (float*)d_out);
}
